// AFAMLayer_43920335569211
// MI455X (gfx1250) — compile-verified, matrix operand order checked by reading, not run on hardware
//
#include <hip/hip_runtime.h>
#include <hip/hip_bf16.h>

#define DI __device__ __forceinline__

typedef __attribute__((ext_vector_type(16))) _Float16 v16h;
typedef __attribute__((ext_vector_type(8)))  _Float16 v8h;
typedef __attribute__((ext_vector_type(8)))  float    v8f;

constexpr int   NB  = 16;     // batch
constexpr int   NC  = 128;    // channels
constexpr int   NHW = 8192;   // H*W
constexpr float EPSN = 1e-5f;

// ---------------- wave reductions (wave32) ----------------
DI float wred32(float x) {
  x += __shfl_xor(x, 1, 32);
  x += __shfl_xor(x, 2, 32);
  x += __shfl_xor(x, 4, 32);
  x += __shfl_xor(x, 8, 32);
  x += __shfl_xor(x, 16, 32);
  return x;
}
DI float wred16(float x) {   // reduces lanes 0-15 and 16-31 independently
  x += __shfl_xor(x, 1, 32);
  x += __shfl_xor(x, 2, 32);
  x += __shfl_xor(x, 4, 32);
  x += __shfl_xor(x, 8, 32);
  return x;
}

// ============================================================
// K1: alpha = sigmoid(fc(cat(Corr,Coh))); agg = Corr - alpha*feats
//     + per-(b,c) instance-stat partials (one 128-h tile per WG)
// grid: (HW/128, B) x 256 threads; dyn LDS = 128*128*4 + 8*128*4
// ============================================================
__global__ __launch_bounds__(256) void k_alpha_agg(
    const float* __restrict__ corr, const float* __restrict__ coh,
    const float* __restrict__ feats, const float* __restrict__ fcw,
    const float* __restrict__ fcb,
    float* __restrict__ agg, float* __restrict__ psum, float* __restrict__ psq)
{
  constexpr int HT = 128;
  extern __shared__ float sm1[];
  float* s_corr = sm1;              // [NC][HT]
  float* s_ap   = sm1 + NC * HT;    // [8][HT] alpha partials per wave

  const int b    = blockIdx.y;
  const int h0   = blockIdx.x * HT;
  const int lane = threadIdx.x & 31;
  const int w    = threadIdx.x >> 5;
  const size_t base = (size_t)b * NC * NHW + h0 + lane * 4;

  float apx = 0.f, apy = 0.f, apz = 0.f, apw = 0.f;
  #pragma unroll 4
  for (int r = 0; r < 16; ++r) {
    const int c = w + 8 * r;
    const float4 c4 = *(const float4*)(corr + base + (size_t)c * NHW);
    const float4 o4 = *(const float4*)(coh  + base + (size_t)c * NHW);
    *(float4*)(s_corr + c * HT + lane * 4) = c4;
    const float wa = fcw[c], wb = fcw[NC + c];
    apx += c4.x * wa + o4.x * wb;
    apy += c4.y * wa + o4.y * wb;
    apz += c4.z * wa + o4.z * wb;
    apw += c4.w * wa + o4.w * wb;
  }
  float4 ap; ap.x = apx; ap.y = apy; ap.z = apz; ap.w = apw;
  *(float4*)(s_ap + w * HT + lane * 4) = ap;
  __syncthreads();

  const float b0 = fcb[0];
  float lx = b0, ly = b0, lz = b0, lw_ = b0;
  #pragma unroll
  for (int wv = 0; wv < 8; ++wv) {
    const float4 p = *(const float4*)(s_ap + wv * HT + lane * 4);
    lx += p.x; ly += p.y; lz += p.z; lw_ += p.w;
  }
  const float ax = 1.f / (1.f + expf(-lx));
  const float ay = 1.f / (1.f + expf(-ly));
  const float az = 1.f / (1.f + expf(-lz));
  const float aw = 1.f / (1.f + expf(-lw_));

  #pragma unroll 4
  for (int r = 0; r < 16; ++r) {
    const int c = w + 8 * r;
    const float4 f4 = *(const float4*)(feats + base + (size_t)c * NHW);
    const float4 c4 = *(const float4*)(s_corr + c * HT + lane * 4);
    float4 a4;
    a4.x = c4.x - ax * f4.x;
    a4.y = c4.y - ay * f4.y;
    a4.z = c4.z - az * f4.z;
    a4.w = c4.w - aw * f4.w;
    *(float4*)(agg + base + (size_t)c * NHW) = a4;
    float s = a4.x + a4.y + a4.z + a4.w;
    float q = a4.x * a4.x + a4.y * a4.y + a4.z * a4.z + a4.w * a4.w;
    s = wred32(s); q = wred32(q);
    if (lane == 0) {
      const int idx = (b * NC + c) * 64 + blockIdx.x;
      psum[idx] = s; psq[idx] = q;
    }
  }
}

// ============================================================
// K2a: reduce per-(b,c) partials -> mean, rstd, var/(var+eps)
// grid: 8 x 256 (2048 threads, one per (b,c))
// ============================================================
__global__ __launch_bounds__(256) void k_fin_a(
    const float* __restrict__ psum, const float* __restrict__ psq, int npart,
    float* __restrict__ m_arr, float* __restrict__ r_arr, float* __restrict__ vt_arr)
{
  const int idx = blockIdx.x * 256 + threadIdx.x;   // < NB*NC
  float s = 0.f, q = 0.f;
  for (int i = 0; i < npart; ++i) { s += psum[idx * npart + i]; q += psq[idx * npart + i]; }
  const float invN = 1.f / (float)NHW;
  const float m = s * invN;
  const float v = fmaxf(q * invN - m * m, 0.f);
  const float r = rsqrtf(v + EPSN);
  m_arr[idx] = m; r_arr[idx] = r; vt_arr[idx] = v * r * r;   // v/(v+eps)
}

// ============================================================
// K2b: closed-form BN over IN'd data -> per-(b,c) scale/shift
// grid: 1 x 128 (one thread per channel)
// ============================================================
__global__ __launch_bounds__(128) void k_fin_b(
    const float* __restrict__ m_arr, const float* __restrict__ r_arr,
    const float* __restrict__ vt_arr, const float* __restrict__ g,
    const float* __restrict__ be, float* __restrict__ scale, float* __restrict__ shift)
{
  const int c = threadIdx.x;
  float vb = 0.f;
  for (int b = 0; b < NB; ++b) vb += vt_arr[b * NC + c];
  vb *= (1.f / (float)NB);
  const float bnr = rsqrtf(vb + EPSN);
  const float gc = g[c] * bnr, bec = be[c];
  for (int b = 0; b < NB; ++b) {
    const int i = b * NC + c;
    const float sc = r_arr[i] * gc;
    scale[i] = sc;
    shift[i] = bec - m_arr[i] * sc;
  }
}

// ============================================================
// Kw: convert both weight matrices to f16
// ============================================================
__global__ __launch_bounds__(256) void k_cvt(
    const float* __restrict__ w1, const float* __restrict__ w2,
    _Float16* __restrict__ w1h, _Float16* __restrict__ w2h)
{
  const int i = blockIdx.x * 256 + threadIdx.x;
  if (i < NC * NC) w1h[i] = (_Float16)w1[i];
  else             w2h[i - NC * NC] = (_Float16)w2[i - NC * NC];
}

// ============================================================
// GEMM: out[b,o,h] = sum_c relu(scale*x + shift) * W[o,c] + bias[o]
// WG = 256 threads (8 waves), tile = 128o x 256h, one b per blockIdx.y
// WMMA f32_16x16x32_f16, K=128 in 4 chunks.
// STATS: fused per-(b,o) sum/sumsq partials for next block's IN.
// dyn LDS = 256*68*4 (X, transposed f16, padded) + 128*136*2 (W f16) = 104448
// ============================================================
DI v16h mk_op(const _Float16* p) {   // lo chunk at p, hi chunk at p+16
  const v8h lo = *(const v8h*)p;
  const v8h hi = *(const v8h*)(p + 16);
  v16h r;
  #pragma unroll
  for (int i = 0; i < 8; ++i) { r[i] = lo[i]; r[i + 8] = hi[i]; }
  return r;
}

template <bool STATS>
__global__ __launch_bounds__(256) void k_gemm(
    const float* __restrict__ x, const _Float16* __restrict__ w16,
    const float* __restrict__ scale, const float* __restrict__ shift,
    const float* __restrict__ bias,
    float* __restrict__ out, float* __restrict__ psum, float* __restrict__ psq)
{
  constexpr int HT  = 256;
  constexpr int LDX = 68;     // dwords per h-row (136 halfs): 68 % 64 == 4
  constexpr int LDW = 136;    // halfs per W row
  extern __shared__ char smraw[];
  unsigned int* sx = (unsigned int*)smraw;                   // [HT][LDX] dwords
  _Float16*     swt = (_Float16*)(smraw + HT * LDX * 4);     // [128][LDW]

  const int b    = blockIdx.y;
  const int ht0  = blockIdx.x * HT;
  const int lane = threadIdx.x & 31;
  const int w    = threadIdx.x >> 5;

  // ---- stage W (f16) into padded LDS ----
  #pragma unroll 4
  for (int i = 0; i < 16; ++i) {
    const int row = i * 8 + w;
    const uint2 d = *(const uint2*)(w16 + row * NC + lane * 4);
    *(uint2*)(swt + row * LDW + lane * 4) = d;
  }

  // ---- stage X: affine + relu + f16, transposed to lds[h][c] ----
  const size_t xbase = (size_t)b * NC * NHW + ht0;
  #pragma unroll 2
  for (int pp = 0; pp < 8; ++pp) {
    const int p = w + 8 * pp;            // c-pair index
    const int c0 = 2 * p, c1 = c0 + 1;
    const float s0 = scale[b * NC + c0], f0 = shift[b * NC + c0];
    const float s1 = scale[b * NC + c1], f1 = shift[b * NC + c1];
    const float* r0 = x + xbase + (size_t)c0 * NHW;
    const float* r1 = x + xbase + (size_t)c1 * NHW;
    #pragma unroll
    for (int j = 0; j < 8; ++j) {
      const int h = lane + 32 * j;
      const float y0 = fmaxf(fmaf(r0[h], s0, f0), 0.f);
      const float y1 = fmaxf(fmaf(r1[h], s1, f1), 0.f);
      union { _Float16 hh[2]; unsigned int u; } pk;
      pk.hh[0] = (_Float16)y0; pk.hh[1] = (_Float16)y1;
      sx[h * LDX + p] = pk.u;
    }
  }
  __syncthreads();

  // ---- per-wave WMMA: 16 o-rows x 256 h ----
  const int o0   = w * 16;
  const int sel  = (lane < 16) ? 0 : 8;
  const int mrow = lane & 15;
  const int grp  = lane >> 4;

  v16h A[4];
  {
    const _Float16* wrow = swt + (o0 + mrow) * LDW;
    #pragma unroll
    for (int kc4 = 0; kc4 < 4; ++kc4) A[kc4] = mk_op(wrow + kc4 * 32 + sel);
  }

  float bsv[8], asum[8], asq[8];
  #pragma unroll
  for (int j = 0; j < 8; ++j) {
    bsv[j] = bias[o0 + j + 8 * grp];
    asum[j] = 0.f; asq[j] = 0.f;
  }

  const size_t obase = (size_t)b * NC * NHW;
  for (int t = 0; t < 16; ++t) {
    const int hh = t * 16;
    const _Float16* xcol = ((const _Float16*)sx) + (hh + mrow) * (LDX * 2);
    v8f acc = {};
    #pragma unroll
    for (int kc4 = 0; kc4 < 4; ++kc4) {
      const v16h Bm = mk_op(xcol + kc4 * 32 + sel);
      acc = __builtin_amdgcn_wmma_f32_16x16x32_f16(
          false, A[kc4], false, Bm, (short)0, acc, false, false);
    }
    const int hcol = ht0 + hh + mrow;
    #pragma unroll
    for (int j = 0; j < 8; ++j) {
      const int o = o0 + j + 8 * grp;
      const float v = acc[j] + bsv[j];
      out[obase + (size_t)o * NHW + hcol] = v;
      if (STATS) { asum[j] += v; asq[j] += v * v; }
    }
  }

  if (STATS) {
    #pragma unroll
    for (int j = 0; j < 8; ++j) {
      const float s = wred16(asum[j]);
      const float q = wred16(asq[j]);
      if ((lane & 15) == 0) {
        const int o = o0 + j + 8 * grp;
        const int idx = (b * NC + o) * 32 + blockIdx.x;
        psum[idx] = s; psq[idx] = q;
      }
    }
  }
}

// ============================================================
extern "C" void kernel_launch(void* const* d_in, const int* in_sizes, int n_in,
                              void* d_out, int out_size, void* d_ws, size_t ws_size,
                              hipStream_t stream) {
  const float* corr  = (const float*)d_in[0];
  const float* coh   = (const float*)d_in[1];
  const float* feats = (const float*)d_in[2];
  const float* fcw   = (const float*)d_in[3];
  const float* fcb   = (const float*)d_in[4];
  const float* w1    = (const float*)d_in[5];
  const float* b1    = (const float*)d_in[6];
  const float* g1    = (const float*)d_in[7];
  const float* be1   = (const float*)d_in[8];
  const float* w2    = (const float*)d_in[9];
  const float* b2    = (const float*)d_in[10];
  const float* g2    = (const float*)d_in[11];
  const float* be2   = (const float*)d_in[12];

  char* ws = (char*)d_ws;
  constexpr size_t TENS = (size_t)NB * NC * NHW * 4;          // 64 MiB
  float*     agg   = (float*)(ws + 0);
  float*     x1    = (float*)(ws + TENS);
  _Float16*  w1h   = (_Float16*)(ws + 2 * TENS);
  _Float16*  w2h   = (_Float16*)(ws + 2 * TENS + 32768);
  float*     psum1 = (float*)(ws + 2 * TENS + 65536);
  float*     psq1  = psum1 + 2048 * 64;
  float*     psum2 = psq1 + 2048 * 64;
  float*     psq2  = psum2 + 2048 * 32;
  float*     m_a   = psq2 + 2048 * 32;
  float*     r_a   = m_a + 2048;
  float*     vt_a  = r_a + 2048;
  float*     sc1   = vt_a + 2048;
  float*     sh1   = sc1 + 2048;
  float*     sc2   = sh1 + 2048;
  float*     sh2   = sc2 + 2048;

  const size_t lds1 = (size_t)NC * 128 * 4 + 8 * 128 * 4;           // 69632
  const size_t ldsg = (size_t)256 * 68 * 4 + (size_t)128 * 136 * 2; // 104448

  // weights -> f16
  k_cvt<<<128, 256, 0, stream>>>(w1, w2, w1h, w2h);

  // alpha + agg + instance-stat partials (block 1)
  k_alpha_agg<<<dim3(NHW / 128, NB), 256, lds1, stream>>>(
      corr, coh, feats, fcw, fcb, agg, psum1, psq1);

  // finalize IN∘BN affine for block 1
  k_fin_a<<<8, 256, 0, stream>>>(psum1, psq1, 64, m_a, r_a, vt_a);
  k_fin_b<<<1, 128, 0, stream>>>(m_a, r_a, vt_a, g1, be1, sc1, sh1);

  // block 1 GEMM (fused norm+relu in, bias+stats out)
  k_gemm<true><<<dim3(NHW / 256, NB), 256, ldsg, stream>>>(
      agg, w1h, sc1, sh1, b1, x1, psum2, psq2);

  // finalize IN∘BN affine for block 2
  k_fin_a<<<8, 256, 0, stream>>>(psum2, psq2, 32, m_a, r_a, vt_a);
  k_fin_b<<<1, 128, 0, stream>>>(m_a, r_a, vt_a, g2, be2, sc2, sh2);

  // block 2 GEMM -> final output
  k_gemm<false><<<dim3(NHW / 256, NB), 256, ldsg, stream>>>(
      x1, w2h, sc2, sh2, b2, (float*)d_out, nullptr, nullptr);
}
